// KOBE_76948634075865
// MI455X (gfx1250) — matrix operand order repaired and checked by reading, not compiled, not run on hardware
//
#include <hip/hip_runtime.h>

typedef __attribute__((ext_vector_type(16))) _Float16 v16h;
typedef __attribute__((ext_vector_type(16))) short    v16s;
typedef __attribute__((ext_vector_type(8)))  float    v8f;

#define NUM_BITS 32
#define NVAR     528
#define WROWS    32      // i index (K dim), 0..31
#define WCOLS    48      // j index (N dim), 0..32 used, 33..47 zero pad

// xor-lane shuffle within each 16-lane group via v_permlane16_b32 (VALU, no DS).
__device__ __forceinline__ float xor_lanes(float v, unsigned s0, unsigned s1) {
    unsigned i = __builtin_bit_cast(unsigned, v);
    unsigned r = __builtin_amdgcn_permlane16(i, i, s0, s1, false, false);
    return __builtin_bit_cast(float, r);
}
// selector tables (nibble per lane: lane j reads lane sel[j] = j^mask)
#define XS8_LO  0xFEDCBA98u
#define XS8_HI  0x76543210u
#define XS4_LO  0x32107654u
#define XS4_HI  0xBA98FEDCu
#define XS2_LO  0x45670123u
#define XS2_HI  0xCDEF89ABu
#define XS1_LO  0x67452301u
#define XS1_HI  0xEFCDAB89u

__global__ __launch_bounds__(256) void ising_order2_wmma(
    const int*   __restrict__ bits,     // [batch][32] 0/1
    const float* __restrict__ vars,     // [528]
    const int*   __restrict__ indices,  // [528][2], sentinel 32
    float*       __restrict__ out,      // [batch]
    int batch)
{
    // W transposed in LDS: Wt[j*32 + i] = W[i][j] (f16).
    __shared__ __align__(32) _Float16 Wt[WCOLS * WROWS];

    const int tid = threadIdx.x;

    // ---- build W once per block ----------------------------------------
    for (int k = tid; k < (WCOLS * WROWS) / 2; k += 256)
        ((int*)Wt)[k] = 0;
    __syncthreads();
    for (int v = tid; v < NVAR; v += 256) {
        int i = indices[2 * v + 0];          // 0..31
        int j = indices[2 * v + 1];          // i+1..32 (sentinel for order-1)
        Wt[j * WROWS + i] = (_Float16)vars[v];
    }
    __syncthreads();

    const int lane  = tid & 31;
    const int nloc  = lane & 15;   // N (col) within tile / M row for A
    const int khalf = lane >> 4;   // half-wave selector

    // ---- hoist B fragments (W is tile-invariant) ------------------------
    // B layout (32x16 f16): lanes 0-15 hold K=0..15, lanes 16-31 K=16..31;
    // element h -> K = khalf*16 + h, col = ntile*16 + nloc.
    v16h B0 = *(const v16h*)&Wt[( 0 + nloc) * WROWS + khalf * 16];
    v16h B1 = *(const v16h*)&Wt[(16 + nloc) * WROWS + khalf * 16];
    v16h B2 = *(const v16h*)&Wt[(32 + nloc) * WROWS + khalf * 16];

    const int num_tiles = batch >> 4;
    const int gwave     = blockIdx.x * 8 + (tid >> 5);
    const int nwaves    = gridDim.x * 8;

    for (int tile = gwave; tile < num_tiles; tile += nwaves) {
        const int m0 = tile << 4;

        // ---- issue all global loads up front ---------------------------
        const int row = m0 + nloc;
        const int4* p0 = (const int4*)(bits + row * NUM_BITS + khalf * 16);
        const int4* p1 = (const int4*)(bits + row * NUM_BITS + khalf * 16 + 8);
        int4 g0 = p0[0], g1 = p0[1];   // K-group low  (8 ints)
        int4 g2 = p1[0], g3 = p1[1];   // K-group high (8 ints)

        int eb0[8], eb1[8];            // epilogue bits (L0/L2 hits)
        #pragma unroll
        for (int r = 0; r < 8; ++r) {
            const int rr = m0 + r + 8 * khalf;
            eb0[r] = bits[rr * NUM_BITS + nloc];
            eb1[r] = bits[rr * NUM_BITS + 16 + nloc];
        }

        // ---- A fragment: 16x32 f16 spins (+1 -> 0x3C00, -1 -> 0xBC00) --
        v16s sv;
        sv[0]  = (short)(0x3C00 | (g0.x << 15));
        sv[1]  = (short)(0x3C00 | (g0.y << 15));
        sv[2]  = (short)(0x3C00 | (g0.z << 15));
        sv[3]  = (short)(0x3C00 | (g0.w << 15));
        sv[4]  = (short)(0x3C00 | (g1.x << 15));
        sv[5]  = (short)(0x3C00 | (g1.y << 15));
        sv[6]  = (short)(0x3C00 | (g1.z << 15));
        sv[7]  = (short)(0x3C00 | (g1.w << 15));
        sv[8]  = (short)(0x3C00 | (g2.x << 15));
        sv[9]  = (short)(0x3C00 | (g2.y << 15));
        sv[10] = (short)(0x3C00 | (g2.z << 15));
        sv[11] = (short)(0x3C00 | (g2.w << 15));
        sv[12] = (short)(0x3C00 | (g3.x << 15));
        sv[13] = (short)(0x3C00 | (g3.y << 15));
        sv[14] = (short)(0x3C00 | (g3.z << 15));
        sv[15] = (short)(0x3C00 | (g3.w << 15));
        v16h A = __builtin_bit_cast(v16h, sv);

        // ---- T = S(16x32) @ W(32x48): 3 WMMAs, single K step -----------
        v8f z = {};
        v8f c0 = __builtin_amdgcn_wmma_f32_16x16x32_f16(false, A, false, B0,
                                                        (short)0, z, false, false);
        v8f c1 = __builtin_amdgcn_wmma_f32_16x16x32_f16(false, A, false, B1,
                                                        (short)0, z, false, false);
        v8f c2 = __builtin_amdgcn_wmma_f32_16x16x32_f16(false, A, false, B2,
                                                        (short)0, z, false, false);

        // ---- epilogue: out[b] = sum_j s[b,j]*T[b,j] + T[b,32] ----------
        // Sign-bit trick: s*T = T xor (bit<<31). Tile2 cols>32 are exact 0.
        float part[8];
        #pragma unroll
        for (int r = 0; r < 8; ++r) {
            const int s0 = eb0[r] << 31;
            const int s1 = eb1[r] << 31;
            float t0 = __builtin_bit_cast(float, __builtin_bit_cast(int, c0[r]) ^ s0);
            float t1 = __builtin_bit_cast(float, __builtin_bit_cast(int, c1[r]) ^ s1);
            part[r] = t0 + t1 + c2[r];
        }

        // Butterfly with stream folding: 8 streams over 16 lanes -> 8 shuffles.
        const bool sel3 = (lane >> 3) & 1;
        #pragma unroll
        for (int k = 0; k < 4; ++k) {
            float keep = sel3 ? part[k + 4] : part[k];
            float send = sel3 ? part[k]     : part[k + 4];
            part[k] = keep + xor_lanes(send, XS8_LO, XS8_HI);
        }
        const bool sel2 = (lane >> 2) & 1;
        #pragma unroll
        for (int k = 0; k < 2; ++k) {
            float keep = sel2 ? part[k + 2] : part[k];
            float send = sel2 ? part[k]     : part[k + 2];
            part[k] = keep + xor_lanes(send, XS4_LO, XS4_HI);
        }
        const bool sel1 = (lane >> 1) & 1;
        {
            float keep = sel1 ? part[1] : part[0];
            float send = sel1 ? part[0] : part[1];
            part[0] = keep + xor_lanes(send, XS2_LO, XS2_HI);
        }
        part[0] += xor_lanes(part[0], XS1_LO, XS1_HI);

        // lane 2r (each half) owns row r: r = (lane>>1)&7
        if ((lane & 1) == 0)
            out[m0 + ((lane >> 1) & 7) + 8 * khalf] = part[0];
    }
}

extern "C" void kernel_launch(void* const* d_in, const int* in_sizes, int n_in,
                              void* d_out, int out_size, void* d_ws, size_t ws_size,
                              hipStream_t stream) {
    const int*   bits    = (const int*)d_in[0];
    const float* vars    = (const float*)d_in[1];
    const int*   indices = (const int*)d_in[2];
    float*       outp    = (float*)d_out;

    const int batch  = in_sizes[0] / NUM_BITS;  // 131072
    const int blocks = 512;                     // 4096 wave32s, 2 tiles each
    hipLaunchKernelGGL(ising_order2_wmma, dim3(blocks), dim3(256), 0, stream,
                       bits, vars, indices, outp, batch);
}
